// SanityNet_73452530696537
// MI455X (gfx1250) — compile-verified, matrix operand order checked by reading, not run on hardware
//
#include <hip/hip_runtime.h>
#include <hip/hip_bf16.h>

// ---- problem constants ----
#define BB   256      // batch
#define FF   1024     // filters
#define NIN  784      // inputs
#define NINP 800      // padded to multiple of 32 (K for fwd, N for stdp)
#define TT   100      // timesteps

#define DECAY 0.9512294245007140f   // exp(-1/20), same for mem/pre/post
#define VTH   16.0f
#define LRP   (0.01f  / 256.0f)     // LR_POST / B
#define LRM   (0.012f / 256.0f)     // LR_PRE  / B

typedef __attribute__((ext_vector_type(8)))  _Float16 v8h;
typedef __attribute__((ext_vector_type(16))) _Float16 v16h;
typedef __attribute__((ext_vector_type(8)))  float    v8f;

// f16 16x32 A / 32x16 B fragment: per lane, two contiguous 8-half (16B) loads.
// lane L: idx = L&15, ks = (L>>4)*8 ; elems 0..7 = K[ks..+7], 8..15 = K[16+ks..+7]
__device__ __forceinline__ v16h load_frag(const _Float16* __restrict__ base) {
  v8h lo = *(const v8h*)(base);
  v8h hi = *(const v8h*)(base + 16);
  v16h r;
#pragma unroll
  for (int i = 0; i < 8; ++i) { r[i] = lo[i]; r[8 + i] = hi[i]; }
  return r;
}

__device__ __forceinline__ v8f wmma_f16(v16h a, v16h b, v8f c) {
  return __builtin_amdgcn_wmma_f32_16x16x32_f16(false, a, false, b, (short)0, c,
                                                false, false);
}

// ---------------- init kernels ----------------
__global__ void init_w_kernel(const float* __restrict__ Win,
                              float* __restrict__ Wf,
                              _Float16* __restrict__ Wh) {
  int idx = blockIdx.x * blockDim.x + threadIdx.x;          // over FF*NINP
  if (idx >= FF * NINP) return;
  int f = idx / NINP, i = idx - f * NINP;
  float w = 0.0f;
  if (i < NIN) { w = Win[f * NIN + i]; Wf[f * NIN + i] = w; }
  Wh[f * NINP + i] = (_Float16)w;
}

__global__ void init_state_kernel(float* __restrict__ mem,
                                  float* __restrict__ apost,
                                  float* __restrict__ cnt,
                                  _Float16* __restrict__ apostT,
                                  float* __restrict__ apre,
                                  _Float16* __restrict__ preT,
                                  _Float16* __restrict__ apreT) {
  int idx = blockIdx.x * blockDim.x + threadIdx.x;          // over BB*FF
  if (idx >= BB * FF) return;
  mem[idx] = 0.0f; apost[idx] = 0.0f; cnt[idx] = 0.0f;
  apostT[idx] = (_Float16)0.0f;
  if (idx < BB * NIN) apre[idx] = 0.0f;
  if (idx < NINP * BB) { preT[idx] = (_Float16)0.0f; apreT[idx] = (_Float16)0.0f; }
}

// ---------------- per-step kernels ----------------
// Poisson encode + apre trace update; emits padded row-major preh and
// transposed (NINP x BB) preT/apreT for the STDP GEMMs (pad rows stay zero).
__global__ void encode_kernel(const float* __restrict__ noise_t,
                              const float* __restrict__ image,
                              float* __restrict__ apre,
                              _Float16* __restrict__ preh,
                              _Float16* __restrict__ preT,
                              _Float16* __restrict__ apreT) {
  int idx = blockIdx.x * blockDim.x + threadIdx.x;          // over BB*NINP
  if (idx >= BB * NINP) return;
  int b = idx / NINP, i = idx - b * NINP;
  float pre = 0.0f;
  if (i < NIN) {
    float p = image[b * NIN + i] * 0.2f;                    // rate*dt
    pre = (noise_t[b * NIN + i] < p) ? 1.0f : 0.0f;
    float a = apre[b * NIN + i] * DECAY + pre;              // apre updated BEFORE dw+
    apre[b * NIN + i] = a;
    preT[i * BB + b]  = (_Float16)pre;
    apreT[i * BB + b] = (_Float16)a;
  }
  preh[b * NINP + i] = (_Float16)pre;
}

// I = pre @ W^T via WMMA, fused LIF update.
// One 16(batch) x 64(filter) strip per wave: A fragment reused across 4 B tiles.
// Fully unrolled K-loop + one-iteration-ahead double buffering.
// waves_per_eu(2): block is 8 waves = exactly 2 waves/SIMD32, so grant the
// allocator the full 512-VGPR budget -> fragments stay live, loads pipeline.
__global__ __launch_bounds__(256)
__attribute__((amdgpu_waves_per_eu(2))) void forward_kernel(
    const _Float16* __restrict__ preh, const _Float16* __restrict__ Wh,
    float* __restrict__ mem, float* __restrict__ cnt,
    _Float16* __restrict__ spkT) {
  int wave = threadIdx.x >> 5, lane = threadIdx.x & 31;
  int tile = blockIdx.x * 8 + wave;                         // 16 x 16 supertiles
  int b0 = (tile >> 4) * 16, f0 = (tile & 15) * 64;
  int mn = lane & 15, h8 = (lane >> 4) * 8;

  const _Float16* rowA = preh + (size_t)(b0 + mn) * NINP + h8;
  const _Float16* rowB = Wh   + (size_t)(f0 + mn) * NINP + h8;

  v8f acc[4] = {};
  v16h aC = load_frag(rowA);
  v16h bC[4];
#pragma unroll
  for (int j = 0; j < 4; ++j) bC[j] = load_frag(rowB + (size_t)j * 16 * NINP);

#pragma unroll
  for (int kk = 0; kk < NINP; kk += 32) {
    v16h aN = aC;
    v16h bN[4] = {bC[0], bC[1], bC[2], bC[3]};
    if (kk + 32 < NINP) {                                   // prefetch next K-slice
      aN = load_frag(rowA + kk + 32);
#pragma unroll
      for (int j = 0; j < 4; ++j)
        bN[j] = load_frag(rowB + (size_t)j * 16 * NINP + kk + 32);
    }
#pragma unroll
    for (int j = 0; j < 4; ++j) acc[j] = wmma_f16(aC, bC[j], acc[j]);
    aC = aN;
#pragma unroll
    for (int j = 0; j < 4; ++j) bC[j] = bN[j];              // SSA renames (free)
  }

  int rbase = b0 + (lane >> 4) * 8;
#pragma unroll
  for (int j = 0; j < 4; ++j) {
    int col = f0 + j * 16 + (lane & 15);
#pragma unroll
    for (int r = 0; r < 8; ++r) {                           // D: VGPR r -> M = r + 8*hi
      int idx = (rbase + r) * FF + col;
      float mv  = mem[idx] * DECAY + acc[j][r];
      float spk = (mv >= VTH) ? 1.0f : 0.0f;
      mem[idx]  = mv * (1.0f - spk);                        // reset on spike
      cnt[idx] += spk;
      spkT[col * BB + (rbase + r)] = (_Float16)spk;
    }
  }
}

// dw+ = spk^T @ apre, dw- = apost^T @ pre (K = batch = 256), fused clipped
// weight update. One 64(filter) x 32(input) supertile per wave:
// 4 B fragments reused across 4 filter subtiles -> 16 wmma / 24 loads per K-step.
// Fully unrolled (8 iterations), shared B fragments double-buffered.
__global__ __launch_bounds__(256)
__attribute__((amdgpu_waves_per_eu(2))) void stdp_kernel(
    const _Float16* __restrict__ spkT, const _Float16* __restrict__ apreT,
    const _Float16* __restrict__ apostT, const _Float16* __restrict__ preT,
    float* __restrict__ Wf, _Float16* __restrict__ Wh) {
  int wave = threadIdx.x >> 5, lane = threadIdx.x & 31;
  int tile = blockIdx.x * 8 + wave;                         // 16 x 25 supertiles
  int mt = tile / 25, nt = tile - mt * 25;
  int f0 = mt * 64, i0 = nt * 32;
  int mn = lane & 15, h8 = (lane >> 4) * 8;

  const _Float16* rAp = spkT   + (size_t)(f0 + mn) * BB + h8;
  const _Float16* rAm = apostT + (size_t)(f0 + mn) * BB + h8;
  const _Float16* rBp = apreT  + (size_t)(i0 + mn) * BB + h8;
  const _Float16* rBm = preT   + (size_t)(i0 + mn) * BB + h8;

  v8f accP[4][2] = {};
  v8f accM[4][2] = {};

  v16h bPC[2], bMC[2];
#pragma unroll
  for (int ii = 0; ii < 2; ++ii) {
    bPC[ii] = load_frag(rBp + (size_t)ii * 16 * BB);
    bMC[ii] = load_frag(rBm + (size_t)ii * 16 * BB);
  }

#pragma unroll
  for (int kk = 0; kk < BB; kk += 32) {
    v16h bPN[2] = {bPC[0], bPC[1]};
    v16h bMN[2] = {bMC[0], bMC[1]};
    if (kk + 32 < BB) {                                     // prefetch next K-slice
#pragma unroll
      for (int ii = 0; ii < 2; ++ii) {
        bPN[ii] = load_frag(rBp + (size_t)ii * 16 * BB + kk + 32);
        bMN[ii] = load_frag(rBm + (size_t)ii * 16 * BB + kk + 32);
      }
    }
#pragma unroll
    for (int j = 0; j < 4; ++j) {
      v16h aP = load_frag(rAp + (size_t)j * 16 * BB + kk);
      v16h aM = load_frag(rAm + (size_t)j * 16 * BB + kk);
#pragma unroll
      for (int ii = 0; ii < 2; ++ii) {
        accP[j][ii] = wmma_f16(aP, bPC[ii], accP[j][ii]);
        accM[j][ii] = wmma_f16(aM, bMC[ii], accM[j][ii]);
      }
    }
#pragma unroll
    for (int ii = 0; ii < 2; ++ii) { bPC[ii] = bPN[ii]; bMC[ii] = bMN[ii]; }
  }

  int fb = f0 + (lane >> 4) * 8;
#pragma unroll
  for (int j = 0; j < 4; ++j) {
#pragma unroll
    for (int ii = 0; ii < 2; ++ii) {
      int i = i0 + ii * 16 + (lane & 15);
      if (i < NIN) {                                        // skip padded columns
#pragma unroll
        for (int r = 0; r < 8; ++r) {
          int f = fb + j * 16 + r;
          float w = Wf[f * NIN + i] + LRP * accP[j][ii][r] - LRM * accM[j][ii][r];
          w = fminf(fmaxf(w, 0.0f), 1.0f);                  // clip [0, W_MAX]
          Wf[f * NIN + i]  = w;
          Wh[f * NINP + i] = (_Float16)w;                   // f16 shadow for next fwd
        }
      }
    }
  }
}

// apost trace update (AFTER weight update, per reference ordering)
__global__ void apost_kernel(const _Float16* __restrict__ spkT,
                             float* __restrict__ apost,
                             _Float16* __restrict__ apostT) {
  int idx = blockIdx.x * blockDim.x + threadIdx.x;          // over BB*FF
  if (idx >= BB * FF) return;
  int b = idx / FF, f = idx - b * FF;
  float a = apost[idx] * DECAY + (float)spkT[f * BB + b];
  apost[idx] = a;
  apostT[f * BB + b] = (_Float16)a;
}

// ---------------- host ----------------
extern "C" void kernel_launch(void* const* d_in, const int* in_sizes, int n_in,
                              void* d_out, int out_size, void* d_ws, size_t ws_size,
                              hipStream_t stream) {
  (void)in_sizes; (void)n_in; (void)out_size; (void)ws_size;
  const float* image = (const float*)d_in[0];   // (B,1,28,28) -> B x 784
  const float* Win   = (const float*)d_in[1];   // (F, 784)
  const float* noise = (const float*)d_in[2];   // (T, B, 784)
  float* cnt = (float*)d_out;                   // (B, F)

  // workspace carve-up (all 16B aligned)
  char* ws = (char*)d_ws;
  float*    Wf     = (float*)(ws);                               // F*NIN f32
  _Float16* Wh     = (_Float16*)(ws + 3211264);                  // F*NINP f16 (padded)
  float*    mem    = (float*)(ws + 4849664);                     // B*F f32
  float*    apre   = (float*)(ws + 5898240);                     // B*NIN f32
  float*    apost  = (float*)(ws + 6701056);                     // B*F f32
  _Float16* apostT = (_Float16*)(ws + 7749632);                  // F*B f16
  _Float16* preh   = (_Float16*)(ws + 8273920);                  // B*NINP f16 (padded)
  _Float16* preT   = (_Float16*)(ws + 8683520);                  // NINP*B f16 (padded)
  _Float16* apreT  = (_Float16*)(ws + 9093120);                  // NINP*B f16 (padded)
  _Float16* spkT   = (_Float16*)(ws + 9502720);                  // F*B f16

  init_w_kernel<<<(FF * NINP + 255) / 256, 256, 0, stream>>>(Win, Wf, Wh);
  init_state_kernel<<<(BB * FF + 255) / 256, 256, 0, stream>>>(
      mem, apost, cnt, apostT, apre, preT, apreT);

  for (int t = 0; t < TT; ++t) {
    const float* noise_t = noise + (size_t)t * BB * NIN;
    encode_kernel<<<(BB * NINP + 255) / 256, 256, 0, stream>>>(
        noise_t, image, apre, preh, preT, apreT);
    // 16 batch-tiles x 16 filter-supertiles = 256 wave tiles, 8 waves/block
    forward_kernel<<<32, 256, 0, stream>>>(preh, Wh, mem, cnt, spkT);
    // 16 filter-supertiles x 25 input-supertiles = 400 wave tiles
    stdp_kernel<<<50, 256, 0, stream>>>(spkT, apreT, apostT, preT, Wf, Wh);
    apost_kernel<<<(BB * FF + 255) / 256, 256, 0, stream>>>(spkT, apost, apostT);
  }
}
